// MultiHeadAttention_25658134626881
// MI455X (gfx1250) — matrix operand order repaired and checked by reading, not compiled, not run on hardware
//
#include <hip/hip_runtime.h>
#include <cstdint>

// ---------------------------------------------------------------------------
// MI455X (gfx1250) multi-head attention forward.
// wave32, WMMA bf16 16x16x32 f32-acc, double-buffered LDS with CDNA5 async
// global->LDS staging (GLOBAL_LOAD_ASYNC_TO_LDS_B128, ASYNCcnt). Memory-bound
// on the 536MB attn output -> 4 passes over attn, everything else via WMMA.
// ---------------------------------------------------------------------------

typedef __attribute__((ext_vector_type(16))) __bf16 v16bf;
typedef __attribute__((ext_vector_type(8)))  __bf16 v8bf;
typedef __attribute__((ext_vector_type(4)))  __bf16 v4bf;
typedef __attribute__((ext_vector_type(8)))  float  v8f;

#define D_MODEL   1024
#define NUM_HEADS 16
#define D_K       64
#define BATCH     2
#define SEQ       2048

// Block tile: 128x64 C per 256-thread block (8 waves as 4x2 grid of 32x32).
#define BM  128
#define BN  64
#define BK  32
#define LDA 40   // BK + 8 pad (bf16): keeps (row*LDA + 8k)*2 16B-aligned
#define LDB 40   // Bt stored transposed [BN][LDB]

// -------------------------- CDNA5 async helpers ----------------------------
// 16B global->LDS async copy; tracked by ASYNCcnt, bypasses VGPRs.
__device__ __forceinline__ void async_copy16(uint32_t lds_off, const void* g) {
  asm volatile("global_load_async_to_lds_b128 %0, %1, off"
               :: "v"(lds_off), "v"((uint64_t)(size_t)g)
               : "memory");
}
__device__ __forceinline__ void wait_async0() {
  asm volatile("s_wait_asynccnt 0x0" ::: "memory");
}
// LDS byte offset of a __shared__ pointer (flat shared addr low 32 bits).
__device__ __forceinline__ uint32_t lds_off(const void* p) {
  return (uint32_t)(size_t)p;
}

__device__ __forceinline__ v8f wmma_bf16(v16bf a, v16bf b, v8f c) {
  return __builtin_amdgcn_wmma_f32_16x16x32_bf16(false, a, false, b, (short)0, c,
                                                 false, false);
}

// A fragment (16x32 bf16, MxK): lane half g holds K [g*8,g*8+8) and [16+g*8,..).
__device__ __forceinline__ v16bf frag_a(const __bf16* As, int mBase, int lane) {
  const int g = lane >> 4;
  const int m = mBase + (lane & 15);
  const v8bf lo = *(const v8bf*)&As[m * LDA + g * 8];
  const v8bf hi = *(const v8bf*)&As[m * LDA + 16 + g * 8];
  v16bf a;
#pragma unroll
  for (int e = 0; e < 8; ++e) { a[e] = lo[e]; a[e + 8] = hi[e]; }
  return a;
}

// B fragment (32x16 bf16, KxN) from LDS tile stored transposed Bt[n][k].
__device__ __forceinline__ v16bf frag_b(const __bf16* Bt, int nBase, int lane) {
  const int g = lane >> 4;
  const int n = nBase + (lane & 15);
  const v8bf lo = *(const v8bf*)&Bt[n * LDB + g * 8];
  const v8bf hi = *(const v8bf*)&Bt[n * LDB + 16 + g * 8];
  v16bf b;
#pragma unroll
  for (int e = 0; e < 8; ++e) { b[e] = lo[e]; b[e + 8] = hi[e]; }
  return b;
}

// ---------------------------------------------------------------------------
// f32 -> bf16 elementwise (activations/weights pre-pass; ~3us of traffic).
// ---------------------------------------------------------------------------
__global__ __launch_bounds__(256) void cvt_bf16_kernel(
    const float* __restrict__ src, __bf16* __restrict__ dst) {
  const int i = blockIdx.x * 256 + threadIdx.x;
  const float4 u = ((const float4*)src)[i];
  v4bf o;
  o[0] = (__bf16)u.x; o[1] = (__bf16)u.y; o[2] = (__bf16)u.z; o[3] = (__bf16)u.w;
  ((v4bf*)dst)[i] = o;
}

// ---------------------------------------------------------------------------
// Y = X @ W^T + b, all-bf16 inputs, async double-buffered staging.
// OUT_MODE 0: f32 [4096,1024]   (final projection)
// OUT_MODE 1: bf16 head-split [B,H,S,DK]   (Q, K)
// OUT_MODE 2: bf16 transposed  [B,H,DK,S]  (V -> makes attn.V staging linear)
// ---------------------------------------------------------------------------
template <int OUT_MODE>
__global__ __launch_bounds__(256) void gemm_xwT_async(
    const __bf16* __restrict__ A, const __bf16* __restrict__ W,
    const float* __restrict__ bias, void* __restrict__ Out) {
  __shared__ __bf16 As[2][BM * LDA];
  __shared__ __bf16 Bt[2][BN * LDB];
  const int K = D_MODEL, N = D_MODEL;
  const int tid = threadIdx.x, lane = tid & 31, w = tid >> 5;
  const int wr = w >> 1, wc = w & 1;
  const int m0 = blockIdx.y * BM, n0 = blockIdx.x * BN;

  const uint32_t asB = lds_off(&As[0][0]);
  const uint32_t btB = lds_off(&Bt[0][0]);
  const int ra = tid >> 2, c8 = (tid & 3) * 8;  // A slots: rows ra, ra+64

  auto issue = [&](int buf, int k0) {
    const uint32_t aO = asB + (uint32_t)(buf * BM * LDA) * 2;
    const uint32_t bO = btB + (uint32_t)(buf * BN * LDB) * 2;
    async_copy16(aO + (uint32_t)(ra * LDA + c8) * 2,
                 A + (size_t)(m0 + ra) * K + k0 + c8);
    async_copy16(aO + (uint32_t)((ra + 64) * LDA + c8) * 2,
                 A + (size_t)(m0 + ra + 64) * K + k0 + c8);
    async_copy16(bO + (uint32_t)(ra * LDB + c8) * 2,       // Bt[n][k] = W[n0+n][k0+k]
                 W + (size_t)(n0 + ra) * K + k0 + c8);
  };

  v8f acc[2][2] = {{{}, {}}, {{}, {}}};

  issue(0, 0);
  wait_async0();
  __syncthreads();
  for (int k0 = 0, it = 0; k0 < K; k0 += BK, ++it) {
    const int cur = it & 1;
    if (k0 + BK < K) issue(cur ^ 1, k0 + BK);  // fill other buffer async
    const __bf16* Ac = As[cur];
    const __bf16* Bc = Bt[cur];
    const v16bf a0 = frag_a(Ac, wr * 32, lane);
    const v16bf a1 = frag_a(Ac, wr * 32 + 16, lane);
    const v16bf b0 = frag_b(Bc, wc * 32, lane);
    const v16bf b1 = frag_b(Bc, wc * 32 + 16, lane);
    acc[0][0] = wmma_bf16(a0, b0, acc[0][0]);
    acc[0][1] = wmma_bf16(a0, b1, acc[0][1]);
    acc[1][0] = wmma_bf16(a1, b0, acc[1][0]);
    acc[1][1] = wmma_bf16(a1, b1, acc[1][1]);
    wait_async0();
    __syncthreads();
  }

  const int g = lane >> 4, nl = lane & 15;
#pragma unroll
  for (int ti = 0; ti < 2; ++ti)
#pragma unroll
    for (int tj = 0; tj < 2; ++tj) {
      const int mT = m0 + wr * 32 + ti * 16;
      const int nT = n0 + wc * 32 + tj * 16;
#pragma unroll
      for (int r = 0; r < 8; ++r) {
        const int row = mT + r + 8 * g;
        const int col = nT + nl;
        const float v = acc[ti][tj][r] + bias[col];
        const int b = row >> 11, s = row & (SEQ - 1);
        const int h = col >> 6, dd = col & 63;
        if (OUT_MODE == 0) {
          ((float*)Out)[(size_t)row * N + col] = v;
        } else if (OUT_MODE == 1) {
          ((__bf16*)Out)[(((size_t)(b * NUM_HEADS + h)) * SEQ + s) * D_K + dd] =
              (__bf16)v;
        } else {  // V^T
          ((__bf16*)Out)[(((size_t)(b * NUM_HEADS + h)) * D_K + dd) * SEQ + s] =
              (__bf16)v;
        }
      }
    }
}

// ---------------------------------------------------------------------------
// scores[bh] = Q[bh] @ K[bh]^T * (1/8), pre-softmax into d_out attn region.
// ---------------------------------------------------------------------------
__global__ __launch_bounds__(256) void scores_kernel(
    const __bf16* __restrict__ Qp, const __bf16* __restrict__ Kp,
    float* __restrict__ attn) {
  __shared__ __bf16 As[2][BM * LDA];
  __shared__ __bf16 Bt[2][BN * LDB];
  const int tid = threadIdx.x, lane = tid & 31, w = tid >> 5;
  const int wr = w >> 1, wc = w & 1;
  const int bh = blockIdx.z;
  const int m0 = blockIdx.y * BM, n0 = blockIdx.x * BN;
  const __bf16* Qh = Qp + (size_t)bh * SEQ * D_K;
  const __bf16* Kh = Kp + (size_t)bh * SEQ * D_K;

  const uint32_t asB = lds_off(&As[0][0]);
  const uint32_t btB = lds_off(&Bt[0][0]);
  const int ra = tid >> 2, c8 = (tid & 3) * 8;

  auto issue = [&](int buf, int k0) {
    const uint32_t aO = asB + (uint32_t)(buf * BM * LDA) * 2;
    const uint32_t bO = btB + (uint32_t)(buf * BN * LDB) * 2;
    async_copy16(aO + (uint32_t)(ra * LDA + c8) * 2,
                 Qh + (size_t)(m0 + ra) * D_K + k0 + c8);
    async_copy16(aO + (uint32_t)((ra + 64) * LDA + c8) * 2,
                 Qh + (size_t)(m0 + ra + 64) * D_K + k0 + c8);
    async_copy16(bO + (uint32_t)(ra * LDB + c8) * 2,
                 Kh + (size_t)(n0 + ra) * D_K + k0 + c8);
  };

  v8f acc[2][2] = {{{}, {}}, {{}, {}}};

  issue(0, 0);
  wait_async0();
  __syncthreads();
#pragma unroll
  for (int k0 = 0, it = 0; k0 < D_K; k0 += BK, ++it) {
    const int cur = it & 1;
    if (k0 + BK < D_K) issue(cur ^ 1, k0 + BK);
    const __bf16* Ac = As[cur];
    const __bf16* Bc = Bt[cur];
    const v16bf a0 = frag_a(Ac, wr * 32, lane);
    const v16bf a1 = frag_a(Ac, wr * 32 + 16, lane);
    const v16bf b0 = frag_b(Bc, wc * 32, lane);
    const v16bf b1 = frag_b(Bc, wc * 32 + 16, lane);
    acc[0][0] = wmma_bf16(a0, b0, acc[0][0]);
    acc[0][1] = wmma_bf16(a0, b1, acc[0][1]);
    acc[1][0] = wmma_bf16(a1, b0, acc[1][0]);
    acc[1][1] = wmma_bf16(a1, b1, acc[1][1]);
    wait_async0();
    __syncthreads();
  }

  float* out = attn + (size_t)bh * SEQ * SEQ;
  const int g = lane >> 4, nl = lane & 15;
#pragma unroll
  for (int ti = 0; ti < 2; ++ti)
#pragma unroll
    for (int tj = 0; tj < 2; ++tj) {
      const int mT = m0 + wr * 32 + ti * 16;
      const int nT = n0 + wc * 32 + tj * 16;
#pragma unroll
      for (int r = 0; r < 8; ++r)
        out[(size_t)(mT + r + 8 * g) * SEQ + (nT + nl)] =
            acc[ti][tj][r] * 0.125f;  // 1/sqrt(d_k)
    }
}

// ---------------------------------------------------------------------------
// In-place row softmax over attn rows of length 2048. One 256-thread block/row.
// ---------------------------------------------------------------------------
__global__ __launch_bounds__(256) void softmax_kernel(float* __restrict__ attn) {
  float* p = attn + (size_t)blockIdx.x * SEQ;
  const int tid = threadIdx.x;
  __shared__ float red[8];

  float v[8];
  float m = -3.0e38f;
#pragma unroll
  for (int i = 0; i < 8; ++i) { v[i] = p[tid + i * 256]; m = fmaxf(m, v[i]); }
#pragma unroll
  for (int off = 16; off; off >>= 1) m = fmaxf(m, __shfl_xor(m, off, 32));
  const int wid = tid >> 5, lane = tid & 31;
  if (lane == 0) red[wid] = m;
  __syncthreads();
  float mm = red[0];
#pragma unroll
  for (int i = 1; i < 8; ++i) mm = fmaxf(mm, red[i]);
  __syncthreads();

  float s = 0.0f;
#pragma unroll
  for (int i = 0; i < 8; ++i) { v[i] = __expf(v[i] - mm); s += v[i]; }
#pragma unroll
  for (int off = 16; off; off >>= 1) s += __shfl_xor(s, off, 32);
  if (lane == 0) red[wid] = s;
  __syncthreads();
  float tot = red[0];
#pragma unroll
  for (int i = 1; i < 8; ++i) tot += red[i];
  const float inv = 1.0f / tot;
#pragma unroll
  for (int i = 0; i < 8; ++i) p[tid + i * 256] = v[i] * inv;
}

// ---------------------------------------------------------------------------
// ctx[b,s,h*64+d] = sum_k attn[bh,s,k] * V[bh,k,d].
// A: attn f32 -> bf16 on VGPR stage (conversion required); B: async from V^T.
// ---------------------------------------------------------------------------
__global__ __launch_bounds__(256) void attnv_kernel(
    const float* __restrict__ attn, const __bf16* __restrict__ Vt,
    __bf16* __restrict__ ctx) {
  __shared__ __bf16 As[2][BM * LDA];
  __shared__ __bf16 Bt[2][BN * LDB];
  const int tid = threadIdx.x, lane = tid & 31, w = tid >> 5;
  const int wr = w >> 1, wc = w & 1;
  const int bh = blockIdx.z, b = bh >> 4, h = bh & 15;
  const int m0 = blockIdx.y * BM;
  const float* Ah = attn + (size_t)bh * SEQ * SEQ;
  const __bf16* Vh = Vt + (size_t)bh * D_K * SEQ;  // [DK, SEQ] row-major

  const uint32_t btB = lds_off(&Bt[0][0]);
  const int ra = tid >> 2, c8 = (tid & 3) * 8;

  auto issueB = [&](int buf, int k0) {  // Bt[n][k] = Vt[n][k0+k], contiguous
    async_copy16(btB + (uint32_t)(buf * BN * LDB + ra * LDB + c8) * 2,
                 Vh + (size_t)ra * SEQ + k0 + c8);
  };
  auto stageA = [&](int buf, int k0) {  // f32 -> bf16 through VGPRs
    const int r = tid >> 3, c4 = (tid & 7) * 4;
#pragma unroll
    for (int p = 0; p < 4; ++p) {
      const int row = p * 32 + r;
      const float4 u = *(const float4*)(Ah + (size_t)(m0 + row) * SEQ + k0 + c4);
      __bf16* d = &As[buf][row * LDA + c4];
      d[0] = (__bf16)u.x; d[1] = (__bf16)u.y;
      d[2] = (__bf16)u.z; d[3] = (__bf16)u.w;
    }
  };

  v8f acc[2][2] = {{{}, {}}, {{}, {}}};

  issueB(0, 0);
  stageA(0, 0);
  wait_async0();
  __syncthreads();
  for (int k0 = 0, it = 0; k0 < SEQ; k0 += BK, ++it) {
    const int cur = it & 1;
    if (k0 + BK < SEQ) {
      issueB(cur ^ 1, k0 + BK);
      stageA(cur ^ 1, k0 + BK);
    }
    const __bf16* Ac = As[cur];
    const __bf16* Bc = Bt[cur];
    const v16bf a0 = frag_a(Ac, wr * 32, lane);
    const v16bf a1 = frag_a(Ac, wr * 32 + 16, lane);
    const v16bf b0 = frag_b(Bc, wc * 32, lane);
    const v16bf b1 = frag_b(Bc, wc * 32 + 16, lane);
    acc[0][0] = wmma_bf16(a0, b0, acc[0][0]);
    acc[0][1] = wmma_bf16(a0, b1, acc[0][1]);
    acc[1][0] = wmma_bf16(a1, b0, acc[1][0]);
    acc[1][1] = wmma_bf16(a1, b1, acc[1][1]);
    wait_async0();
    __syncthreads();
  }

  const int g = lane >> 4, nl = lane & 15;
#pragma unroll
  for (int ti = 0; ti < 2; ++ti)
#pragma unroll
    for (int tj = 0; tj < 2; ++tj) {
      const int mT = m0 + wr * 32 + ti * 16;
      const int nT = wc * 32 + tj * 16;  // head-local column (0..63)
#pragma unroll
      for (int r = 0; r < 8; ++r) {
        const int s = mT + r + 8 * g;
        ctx[((size_t)b * SEQ + s) * D_MODEL + h * D_K + nT + nl] =
            (__bf16)acc[ti][tj][r];
      }
    }
}

// ---------------------------------------------------------------------------
extern "C" void kernel_launch(void* const* d_in, const int* in_sizes, int n_in,
                              void* d_out, int out_size, void* d_ws,
                              size_t ws_size, hipStream_t stream) {
  (void)in_sizes; (void)n_in; (void)out_size; (void)ws_size;
  const float* query = (const float*)d_in[0];
  const float* key_i = (const float*)d_in[1];
  const float* value = (const float*)d_in[2];
  const float* W_q = (const float*)d_in[3];
  const float* b_q = (const float*)d_in[4];
  const float* W_k = (const float*)d_in[5];
  const float* b_k = (const float*)d_in[6];
  const float* W_v = (const float*)d_in[7];
  const float* b_v = (const float*)d_in[8];
  const float* W_o = (const float*)d_in[9];
  const float* b_o = (const float*)d_in[10];

  float* outO = (float*)d_out;                              // [B,S,D] f32
  float* attn = outO + (size_t)BATCH * SEQ * D_MODEL;       // [B,H,S,S] f32

  const size_t NA = (size_t)BATCH * SEQ * D_MODEL;          // 4,194,304
  const size_t NW = (size_t)D_MODEL * D_MODEL;              // 1,048,576
  __bf16* Xq  = (__bf16*)d_ws;       // bf16 activations
  __bf16* Xk  = Xq + NA;
  __bf16* Xv  = Xk + NA;
  __bf16* Wqb = Xv + NA;             // bf16 weights
  __bf16* Wkb = Wqb + NW;
  __bf16* Wvb = Wkb + NW;
  __bf16* Wob = Wvb + NW;
  __bf16* Qp  = Wob + NW;            // [B,H,S,DK]
  __bf16* Kp  = Qp + NA;
  __bf16* Vt  = Kp + NA;             // [B,H,DK,S]
  __bf16* ctx = Vt + NA;             // [B,S,D]   (total ws ~67 MB)

  const dim3 blk(256);
  cvt_bf16_kernel<<<dim3(NA / 1024), blk, 0, stream>>>(query, Xq);
  cvt_bf16_kernel<<<dim3(NA / 1024), blk, 0, stream>>>(key_i, Xk);
  cvt_bf16_kernel<<<dim3(NA / 1024), blk, 0, stream>>>(value, Xv);
  cvt_bf16_kernel<<<dim3(NW / 1024), blk, 0, stream>>>(W_q, Wqb);
  cvt_bf16_kernel<<<dim3(NW / 1024), blk, 0, stream>>>(W_k, Wkb);
  cvt_bf16_kernel<<<dim3(NW / 1024), blk, 0, stream>>>(W_v, Wvb);
  cvt_bf16_kernel<<<dim3(NW / 1024), blk, 0, stream>>>(W_o, Wob);

  const dim3 gProj(D_MODEL / BN, (BATCH * SEQ) / BM);       // (16, 32)
  gemm_xwT_async<1><<<gProj, blk, 0, stream>>>(Xq, Wqb, b_q, Qp);
  gemm_xwT_async<1><<<gProj, blk, 0, stream>>>(Xk, Wkb, b_k, Kp);
  gemm_xwT_async<2><<<gProj, blk, 0, stream>>>(Xv, Wvb, b_v, Vt);

  const dim3 gSc(SEQ / BN, SEQ / BM, BATCH * NUM_HEADS);    // (32, 16, 32)
  scores_kernel<<<gSc, blk, 0, stream>>>(Qp, Kp, attn);

  softmax_kernel<<<dim3(BATCH * NUM_HEADS * SEQ), blk, 0, stream>>>(attn);

  const dim3 gAV(1, SEQ / BM, BATCH * NUM_HEADS);           // (1, 16, 32)
  attnv_kernel<<<gAV, blk, 0, stream>>>(attn, Vt, ctx);

  gemm_xwT_async<0><<<gProj, blk, 0, stream>>>(ctx, Wob, b_o, outO);
}
